// MoE_Cascaded_80960133529778
// MI455X (gfx1250) — compile-verified, matrix operand order checked by reading, not run on hardware
//
#include <hip/hip_runtime.h>
#include <hip/hip_bf16.h>

typedef __attribute__((ext_vector_type(16))) __bf16 v16bf;
typedef __attribute__((ext_vector_type(8)))  float  v8f;

constexpr int T = 8192, M = 1024, H = 4096, E = 8, CAP = 2560;

// ---------------------------------------------------------------- converts
__global__ void cvt_kernel(const float* __restrict__ s, __bf16* __restrict__ d, int n4) {
    int i = blockIdx.x * 256 + threadIdx.x;
    if (i < n4) {
        float4 v = ((const float4*)s)[i];
        d[4 * i + 0] = (__bf16)v.x;
        d[4 * i + 1] = (__bf16)v.y;
        d[4 * i + 2] = (__bf16)v.z;
        d[4 * i + 3] = (__bf16)v.w;
    }
}

// fc2_w [E,H,M] f32  ->  w2t [E,M,H] bf16  (make K=H contiguous for GEMM2)
__global__ void transpose_w2_kernel(const float* __restrict__ w2, __bf16* __restrict__ w2t) {
    __shared__ float tile[32][33];
    int e = blockIdx.z;
    int h0 = blockIdx.x * 32, m0 = blockIdx.y * 32;
    const float* src = w2 + (size_t)e * H * M;
    for (int i = threadIdx.y; i < 32; i += 8)
        tile[i][threadIdx.x] = src[(size_t)(h0 + i) * M + m0 + threadIdx.x];
    __syncthreads();
    __bf16* dst = w2t + (size_t)e * M * H;
    for (int i = threadIdx.y; i < 32; i += 8)
        dst[(size_t)(m0 + i) * H + h0 + threadIdx.x] = (__bf16)tile[threadIdx.x][i];
}

// ---------------------------------------------------------------- gating
__global__ void gate_kernel(const float* __restrict__ x, const float* __restrict__ gw,
                            int* __restrict__ ek, float* __restrict__ gk) {
    int wave = threadIdx.x >> 5, lane = threadIdx.x & 31;
    int t = blockIdx.x * 8 + wave;
    float acc[E];
#pragma unroll
    for (int e = 0; e < E; ++e) acc[e] = 0.f;
    const float* xr = x + (size_t)t * M;
    for (int m = lane; m < M; m += 32) {
        float xv = xr[m];
        const float* g = gw + (size_t)m * E;
#pragma unroll
        for (int e = 0; e < E; ++e) acc[e] += xv * g[e];
    }
#pragma unroll
    for (int off = 16; off > 0; off >>= 1)
#pragma unroll
        for (int e = 0; e < E; ++e) acc[e] += __shfl_xor(acc[e], off, 32);
    if (lane == 0) {
        float mx = acc[0];
#pragma unroll
        for (int e = 1; e < E; ++e) mx = fmaxf(mx, acc[e]);
        float p[E];
#pragma unroll
        for (int e = 0; e < E; ++e) p[e] = __expf(acc[e] - mx);
        int e0 = 0;
#pragma unroll
        for (int e = 1; e < E; ++e) if (p[e] > p[e0]) e0 = e;
        int e1 = (e0 == 0) ? 1 : 0;
#pragma unroll
        for (int e = 0; e < E; ++e) if (e != e0 && p[e] > p[e1]) e1 = e;
        float s = p[e0] + p[e1];
        ek[t]     = e0;  ek[T + t] = e1;
        gk[t]     = p[e0] / s;
        gk[T + t] = p[e1] / s;
    }
}

// single-wave ordered scan, k-major (all k=0 entries before k=1), GShard capacity
__global__ void scan_kernel(const int* __restrict__ ek, float* __restrict__ gk,
                            int* __restrict__ slots, int* __restrict__ rowtok,
                            int* __restrict__ counts) {
    __shared__ int cnt[E];
    int lane = threadIdx.x;
    if (lane < E) cnt[lane] = 0;
    __syncthreads();
    for (int k = 0; k < 2; ++k) {
        for (int base = 0; base < T; base += 32) {
            int t = base + lane;
            int e = ek[k * T + t];
            unsigned eqmask = 0;
#pragma unroll
            for (int ee = 0; ee < E; ++ee) {
                unsigned b = __builtin_amdgcn_ballot_w32(e == ee);
                if (e == ee) eqmask = b;
            }
            unsigned lt = (1u << lane) - 1u;
            int prefix = __popc(eqmask & lt);
            int nmatch = __popc(eqmask);
            int base_e = cnt[e];
            int pos = base_e + prefix;
            bool valid = pos < CAP;
            int slot = valid ? e * CAP + pos : -1;
            slots[k * T + t] = slot;
            if (valid) rowtok[slot] = t;
            else       gk[k * T + t] = 0.f;
            if (prefix == nmatch - 1) cnt[e] = base_e + nmatch;  // leader lane
            __syncthreads();
        }
    }
    if (lane < E) counts[lane] = min(cnt[lane], CAP);
}

// ---------------------------------------------------------------- WMMA GEMM
#define TM 128
#define TN 128
#define TK 64
#define LDT 72  // row pitch in bf16 elems: 144B, keeps 16B alignment, avoids conflicts

__device__ inline v16bf load_frag(const __bf16* base, int row, int k0, int lane) {
    int r  = row + (lane & 15);
    int kb = k0 + ((lane >> 4) << 3);      // lanes 0-15: K 0-7 & 16-23; lanes 16-31: +8
    const __bf16* p = base + r * LDT + kb;
    union { v16bf v; uint4 q[2]; } f;
    f.q[0] = *(const uint4*)p;
    f.q[1] = *(const uint4*)(p + 16);
    return f.v;
}

// NT GEMM: A [rows,K] K-contig (optionally row-gathered), B [N,K] K-contig per expert.
// LDS double-buffered: stage of tile kt+1 overlaps other waves' WMMA of tile kt;
// staging registers are transient (load->wait->ds_store) so no spills under the
// CU-mode 128-VGPR/wave budget for 256-thread blocks.
// RELU_BF16=1 -> out_bf = relu(acc+bias) as bf16 ; else out_f = acc+bias as f32.
template <bool RELU_BF16>
__global__ __launch_bounds__(256) void gemm_kernel(
        const __bf16* __restrict__ A, int lda, int a_gather,
        const int* __restrict__ rowtok,
        const __bf16* __restrict__ B, const float* __restrict__ bias,
        const int* __restrict__ counts, int N, int Kdim,
        __bf16* __restrict__ out_bf, float* __restrict__ out_f) {
    __shared__ __bf16 sA[2][TM * LDT];
    __shared__ __bf16 sB[2][TN * LDT];
    int e = blockIdx.z;
    int n_e = counts[e];
    int row0 = blockIdx.y * TM;
    if (row0 >= n_e) return;
    int col0 = blockIdx.x * TN;
    int tid = threadIdx.x, lane = tid & 31, wave = tid >> 5;
    int wrow = (wave >> 2) * 64;   // 0 / 64
    int wcol = (wave & 3) * 32;    // 0 / 32 / 64 / 96

    const __bf16* Bex = B + (size_t)e * N * Kdim;
    const float*  bse = bias + (size_t)e * N;

    // precompute per-thread copy slots: 4 x 16B chunks for A, 4 for B
    const __bf16* pA[4];
    const __bf16* pB[4];
    int dA[4], dB[4];
#pragma unroll
    for (int it = 0; it < 4; ++it) {
        int i = tid + it * 256;
        int r = i >> 3, c = (i & 7) << 3;
        dA[it] = r * LDT + c;
        dB[it] = r * LDT + c;
        size_t srow;
        if (a_gather) {
            int grow = row0 + r;
            int tok = (grow < n_e) ? rowtok[e * CAP + grow] : 0;
            srow = (size_t)tok;
        } else {
            srow = (size_t)(e * CAP + row0 + r);
        }
        pA[it] = A + srow * (size_t)lda + c;
        pB[it] = Bex + (size_t)(col0 + r) * Kdim + c;
    }

    // stage one K-tile into LDS buffer `buf` (registers here are transient)
    auto stage = [&](int k, int buf) {
        uint4 a0 = *(const uint4*)(pA[0] + k);
        uint4 a1 = *(const uint4*)(pA[1] + k);
        uint4 a2 = *(const uint4*)(pA[2] + k);
        uint4 a3 = *(const uint4*)(pA[3] + k);
        uint4 b0 = *(const uint4*)(pB[0] + k);
        uint4 b1 = *(const uint4*)(pB[1] + k);
        uint4 b2 = *(const uint4*)(pB[2] + k);
        uint4 b3 = *(const uint4*)(pB[3] + k);
        *(uint4*)&sA[buf][dA[0]] = a0;
        *(uint4*)&sA[buf][dA[1]] = a1;
        *(uint4*)&sA[buf][dA[2]] = a2;
        *(uint4*)&sA[buf][dA[3]] = a3;
        *(uint4*)&sB[buf][dB[0]] = b0;
        *(uint4*)&sB[buf][dB[1]] = b1;
        *(uint4*)&sB[buf][dB[2]] = b2;
        *(uint4*)&sB[buf][dB[3]] = b3;
    };

    v8f zero = {};
    v8f acc[4][2];
#pragma unroll
    for (int i = 0; i < 4; ++i)
#pragma unroll
        for (int j = 0; j < 2; ++j) acc[i][j] = zero;

    stage(0, 0);
    int cur = 0;
    int ntiles = Kdim / TK;
    for (int kt = 0; kt < ntiles; ++kt) {
        __syncthreads();                       // buf[cur] visible; alt buffer free
        if (kt + 1 < ntiles) stage((kt + 1) * TK, cur ^ 1);
        const __bf16* cA = &sA[cur][0];
        const __bf16* cB = &sB[cur][0];
#pragma unroll
        for (int ks = 0; ks < TK; ks += 32) {
            v16bf afrag[4], bfrag[2];
#pragma unroll
            for (int i = 0; i < 4; ++i) afrag[i] = load_frag(cA, wrow + i * 16, ks, lane);
#pragma unroll
            for (int j = 0; j < 2; ++j) bfrag[j] = load_frag(cB, wcol + j * 16, ks, lane);
#pragma unroll
            for (int i = 0; i < 4; ++i)
#pragma unroll
                for (int j = 0; j < 2; ++j)
                    acc[i][j] = __builtin_amdgcn_wmma_f32_16x16x32_bf16(
                        false, afrag[i], false, bfrag[j], (short)0, acc[i][j], false, false);
        }
        cur ^= 1;
    }

    int rsel = (lane >> 4) << 3;   // C/D layout: lanes 16-31 hold rows +8
    int csel = lane & 15;
#pragma unroll
    for (int i = 0; i < 4; ++i)
#pragma unroll
        for (int j = 0; j < 2; ++j) {
            int baser = row0 + wrow + i * 16 + rsel;
            int basec = col0 + wcol + j * 16 + csel;
            float bv = bse[basec];
#pragma unroll
            for (int v = 0; v < 8; ++v) {
                int rr = baser + v;
                if (rr < n_e) {
                    float val = acc[i][j][v] + bv;
                    if (RELU_BF16) {
                        val = fmaxf(val, 0.f);
                        out_bf[(size_t)(e * CAP + rr) * N + basec] = (__bf16)val;
                    } else {
                        out_f[(size_t)(e * CAP + rr) * N + basec] = val;
                    }
                }
            }
        }
}

// ---------------------------------------------------------------- combine
__global__ void combine_kernel(const float* __restrict__ y, const int* __restrict__ slots,
                               const float* __restrict__ gk, float* __restrict__ out) {
    size_t idx = (size_t)blockIdx.x * 256 + threadIdx.x;
    int t = (int)(idx >> 10);
    int m = (int)(idx & 1023);
    int s0 = slots[t], s1 = slots[T + t];
    float v = 0.f;
    if (s0 >= 0) v += gk[t]     * y[(size_t)s0 * M + m];
    if (s1 >= 0) v += gk[T + t] * y[(size_t)s1 * M + m];
    out[idx] = v;
}

// ---------------------------------------------------------------- launch
extern "C" void kernel_launch(void* const* d_in, const int* in_sizes, int n_in,
                              void* d_out, int out_size, void* d_ws, size_t ws_size,
                              hipStream_t stream) {
    const float* x      = (const float*)d_in[0];
    const float* gate_w = (const float*)d_in[1];
    const float* fc1_w  = (const float*)d_in[2];
    const float* fc1_b  = (const float*)d_in[3];
    const float* fc2_w  = (const float*)d_in[4];
    const float* fc2_b  = (const float*)d_in[5];
    float* out = (float*)d_out;

    char* ws = (char*)d_ws;
    size_t off = 0;
    auto take = [&](size_t bytes) -> char* {
        char* p = ws + off;
        off += (bytes + 255) & ~(size_t)255;
        return p;
    };
    __bf16* xb     = (__bf16*)take((size_t)T * M * 2);
    __bf16* w1b    = (__bf16*)take((size_t)E * H * M * 2);
    __bf16* w2t    = (__bf16*)take((size_t)E * M * H * 2);
    __bf16* hbuf   = (__bf16*)take((size_t)E * CAP * H * 2);
    float*  ybuf   = (float*)take((size_t)E * CAP * M * 4);
    int*    ek     = (int*)take((size_t)2 * T * 4);
    float*  gk     = (float*)take((size_t)2 * T * 4);
    int*    slots  = (int*)take((size_t)2 * T * 4);
    int*    rowtok = (int*)take((size_t)E * CAP * 4);
    int*    counts = (int*)take((size_t)64);

    // precision conversions / weight transpose
    cvt_kernel<<<(T * M / 4 + 255) / 256, 256, 0, stream>>>(x, xb, T * M / 4);
    cvt_kernel<<<((size_t)E * H * M / 4 + 255) / 256, 256, 0, stream>>>(fc1_w, w1b, E * H * M / 4);
    transpose_w2_kernel<<<dim3(H / 32, M / 32, E), dim3(32, 8), 0, stream>>>(fc2_w, w2t);

    // gating + ordered capacity scan
    gate_kernel<<<T / 8, 256, 0, stream>>>(x, gate_w, ek, gk);
    scan_kernel<<<1, 32, 0, stream>>>(ek, gk, slots, rowtok, counts);

    // fc1: gather X rows, [C,1024] @ W1[4096,1024] (NT), relu -> bf16 h
    gemm_kernel<true><<<dim3(H / TN, CAP / TM, E), 256, 0, stream>>>(
        xb, M, 1, rowtok, w1b, fc1_b, counts, H, M, hbuf, nullptr);

    // fc2: [C,4096] @ w2t[1024,4096] (NT) -> f32 y
    gemm_kernel<false><<<dim3(M / TN, CAP / TM, E), 256, 0, stream>>>(
        hbuf, H, 0, nullptr, w2t, fc2_b, counts, M, H, nullptr, ybuf);

    // weighted gather back to [T,M]
    combine_kernel<<<(int)(((size_t)T * M + 255) / 256), 256, 0, stream>>>(ybuf, slots, gk, out);
}
